// MultiHeadAttention_61976378081333
// MI455X (gfx1250) — matrix operand order repaired and checked by reading, not compiled, not run on hardware
//
#include <hip/hip_runtime.h>
#include <hip/hip_bf16.h>
#include <stdint.h>

typedef _Float16 v16h __attribute__((ext_vector_type(16)));
typedef float    v8f  __attribute__((ext_vector_type(8)));

#define B_   2
#define T_   4096
#define HID_ 512
#define H_   8
#define DH_  64

__device__ __forceinline__ v8f wmma_f16(v16h a, v16h b, v8f c) {
  // D = A(16x32 f16) * B(32x16 f16) + C(16x16 f32)
  return __builtin_amdgcn_wmma_f32_16x16x32_f16(
      /*neg_a=*/false, a, /*neg_b=*/false, b,
      /*c_mod=*/(short)0, c, /*reuse_a=*/false, /*reuse_b=*/false);
}

// Async copy 16B global -> LDS (CDNA5 GLOBAL_LOAD_ASYNC_TO_LDS_B128, ASYNCcnt).
__device__ __forceinline__ void async_b128(const _Float16* g, _Float16* l) {
  const uint32_t lds_addr = (uint32_t)(uintptr_t)l;  // low 32 bits = LDS offset
  asm volatile("global_load_async_to_lds_b128 %0, %1, off"
               :: "v"(lds_addr), "v"(g) : "memory");
}

// ds_swizzle xor-shuffle within wave32 (group-of-32 mode: and=0x1f, xor=XMASK)
template <int XMASK>
__device__ __forceinline__ float swz_xor(float x) {
  constexpr int imm = (XMASK << 10) | 0x1F;
  return __builtin_bit_cast(
      float, __builtin_amdgcn_ds_swizzle(__builtin_bit_cast(int, x), imm));
}

// ---------------------------------------------------------------------------
// f32 -> f16 pack (one pass; keeps all GEMM hot loops conversion-free)
// ---------------------------------------------------------------------------
__global__ void __launch_bounds__(256)
cvt_f16_kernel(const float* __restrict__ in, _Float16* __restrict__ out, int n) {
  const int i = (blockIdx.x * blockDim.x + threadIdx.x) * 4;
  if (i >= n) return;
  const float4 v = *(const float4*)(in + i);
  out[i + 0] = (_Float16)v.x;
  out[i + 1] = (_Float16)v.y;
  out[i + 2] = (_Float16)v.z;
  out[i + 3] = (_Float16)v.w;
}

// ---------------------------------------------------------------------------
// GEMM: acc = X[N,512](f16) * W^T(f16) + bias.
// Each wave computes 16 rows x 64 cols (4 accumulators): the A fragment is
// loaded once per K-step and reused by 4 WMMAs; all 4 waves of a block share
// the same 64 W columns (WGP$ hits). block = 128, grid = (N/64, HID/64).
// Epilogues:
//   MODE 0: Yf[row,col] = acc + bias                     (f32, final Wo proj)
//   MODE 1: RoPE + DH^-0.25 scale -> Yh [B,H,T,DH] f16   (Q / K proj;
//           pair (d, d+32) = acc[t] / acc[t+2] in the SAME lane -> no LDS)
//   MODE 2: transpose             -> Yh [B,H,DH,T] f16   (V proj)
// ---------------------------------------------------------------------------
template <int MODE>
__global__ void __launch_bounds__(128)
gemm512_kernel(const _Float16* __restrict__ X, const _Float16* __restrict__ W,
               const float* __restrict__ bias,
               float* __restrict__ Yf, _Float16* __restrict__ Yh) {
  const int wave = threadIdx.x >> 5;
  const int lane = threadIdx.x & 31;
  const int hlf  = lane >> 4;
  const int ln   = lane & 15;
  const int row0 = blockIdx.x * 64 + wave * 16;
  const int col0 = blockIdx.y * 64;

  const _Float16* xp = X + (size_t)(row0 + ln) * HID_;

  v8f acc[4];
  {
    v8f z = {};
    #pragma unroll
    for (int t = 0; t < 4; ++t) acc[t] = z;
  }

  #pragma unroll 4
  for (int k = 0; k < HID_; k += 32) {
    v16h a;
    const _Float16* pa = xp + k + hlf * 16;  // A: row=ln, dims {d0..d0+15}
    #pragma unroll
    for (int j = 0; j < 8; ++j) { a[j] = pa[j]; a[8 + j] = pa[8 + j]; }
    #pragma unroll
    for (int t = 0; t < 4; ++t) {
      v16h bf;  // B: col = col0+t*16+ln, 16 contiguous k values
      const _Float16* pb = W + (size_t)(col0 + t * 16 + ln) * HID_ + k + hlf * 16;
      #pragma unroll
      for (int e = 0; e < 16; ++e) bf[e] = pb[e];
      acc[t] = wmma_f16(a, bf, acc[t]);
    }
  }

  if constexpr (MODE == 0) {
    #pragma unroll
    for (int t = 0; t < 4; ++t) {
      const int col = col0 + t * 16 + ln;
      const float bv = bias[col];
      #pragma unroll
      for (int v = 0; v < 8; ++v) {
        const int row = row0 + v + 8 * hlf;
        Yf[(size_t)row * HID_ + col] = acc[t][v] + bv;
      }
    }
  } else if constexpr (MODE == 1) {
    // RoPE fully in registers: d = t*16+ln pairs with d+32 = (t+2)*16+ln
    const int hh = blockIdx.y;
    const int b  = row0 >> 12;          // 64 | 4096 -> whole tile shares b
    const int tb = row0 & (T_ - 1);
    float bl[4];
    #pragma unroll
    for (int t = 0; t < 4; ++t) bl[t] = bias[col0 + t * 16 + ln];
    #pragma unroll
    for (int t = 0; t < 2; ++t) {
      const int d = t * 16 + ln;        // d < 32, rotation index i == d
      const float inv_freq = __expf(-(float)d * 0.28782313662425576f); // 1e4^(-d/32)
      #pragma unroll
      for (int v = 0; v < 8; ++v) {
        const int tok = tb + v + 8 * hlf;
        float s, c;
        __sincosf((float)tok * inv_freq, &s, &c);
        const float x1 = acc[t][v] + bl[t];
        const float x2 = acc[t + 2][v] + bl[t + 2];
        const size_t base = ((size_t)(b * H_ + hh) * T_ + tok) * DH_;
        Yh[base + d]      = (_Float16)((x1 * c - x2 * s) * 0.35355339059327373f);
        Yh[base + d + 32] = (_Float16)((x1 * s + x2 * c) * 0.35355339059327373f);
      }
    }
  } else {  // MODE 2: V transposed [B,H,DH,T]
    const int b  = row0 >> 12;
    const int tb = row0 & (T_ - 1);
    #pragma unroll
    for (int t = 0; t < 4; ++t) {
      const int d = t * 16 + ln;
      const float bv = bias[col0 + d];
      _Float16* dst = Yh + ((size_t)(b * H_ + blockIdx.y) * DH_ + d) * T_ + tb;
      #pragma unroll
      for (int v = 0; v < 8; ++v)
        dst[v + 8 * hlf] = (_Float16)(acc[t][v] + bv);
    }
  }
}

// ---------------------------------------------------------------------------
// Flash attention: block = 128 (4 waves), 16 query rows/block, grid=(T/16,B*H).
// Waves split keys (split-K online softmax). Each wave double-buffers its own
// 32-key K tile (4KB) + V tile (4KB) in LDS via async-to-LDS b128 copies.
// ---------------------------------------------------------------------------
__global__ void __launch_bounds__(128)
flash_attn_kernel(const _Float16* __restrict__ qh, const _Float16* __restrict__ kh,
                  const _Float16* __restrict__ vt, const float* __restrict__ mask,
                  _Float16* __restrict__ ctx) {
  __shared__ _Float16 lds_k[4][2][32 * 64];  // [wave][buf][key][dim]
  __shared__ _Float16 lds_v[4][2][64 * 32];  // [wave][buf][dim][key]
  __shared__ _Float16 lds_p[4][16 * 32];     // P staging (C-layout -> A-layout)
  __shared__ float    lds_m[4][16];
  __shared__ float    lds_l[4][16];
  __shared__ float    lds_o[4][16][64];

  const int b    = blockIdx.y >> 3;
  const int h    = blockIdx.y & (H_ - 1);
  const int q0   = blockIdx.x * 16;
  const int wave = threadIdx.x >> 5;
  const int lane = threadIdx.x & 31;
  const int hlf  = lane >> 4;
  const int ln   = lane & 15;

  const _Float16* Qb = qh + ((size_t)(b * H_ + h) * T_ + q0) * DH_;
  const _Float16* Kb = kh + (size_t)(b * H_ + h) * T_ * DH_;
  const _Float16* Vb = vt + (size_t)(b * H_ + h) * DH_ * T_;
  const float*    mb = mask + (size_t)b * T_;

  // stage this wave's 32-key K/V chunk into LDS buffer `buf` (16 async b128/wave)
  auto stage = [&](int s0, int buf) {
    const _Float16* kg = Kb + (size_t)(s0 + lane) * DH_;   // one 128B key row / lane
    _Float16* kl = &lds_k[wave][buf][lane * DH_];
    #pragma unroll
    for (int j = 0; j < 8; ++j) async_b128(kg + j * 8, kl + j * 8);
    #pragma unroll
    for (int r = 0; r < 2; ++r) {                          // two 64B dim rows / lane
      const int d = lane * 2 + r;
      const _Float16* vg = Vb + (size_t)d * T_ + s0;
      _Float16* vl = &lds_v[wave][buf][d * 32];
      #pragma unroll
      for (int j = 0; j < 4; ++j) async_b128(vg + j * 8, vl + j * 8);
    }
  };

  // Q A-fragments (held in registers for the whole kernel)
  v16h qa[2];
  #pragma unroll
  for (int f = 0; f < 2; ++f) {
    const _Float16* p = Qb + (size_t)ln * DH_ + f * 32 + hlf * 16;
    #pragma unroll
    for (int j = 0; j < 8; ++j) { qa[f][j] = p[j]; qa[f][8 + j] = p[8 + j]; }
  }

  float m[8], l[8];
  #pragma unroll
  for (int v = 0; v < 8; ++v) { m[v] = -1e30f; l[v] = 0.f; }
  v8f accO[4];
  #pragma unroll
  for (int t = 0; t < 4; ++t) { v8f z = {}; accO[t] = z; }

  const int NITER = T_ / (4 * 32);  // 32 iterations, 32 keys/wave/iter
  stage(wave * 32, 0);              // preload chunk 0

  for (int it = 0; it < NITER; ++it) {
    const int cur = it & 1;
    const int s0 = (it * 4 + wave) * 32;

    if (it + 1 < NITER) {
      // previous reads of the other buffer must have landed before overwrite
      asm volatile("s_wait_dscnt 0" ::: "memory");
      stage(((it + 1) * 4 + wave) * 32, cur ^ 1);
      // 16 newer async ops in flight => the 16 for `cur` are complete (in-order)
      asm volatile("s_wait_asynccnt 0x10" ::: "memory");
    } else {
      asm volatile("s_wait_asynccnt 0x0" ::: "memory");
    }

    const _Float16* kbuf = &lds_k[wave][cur][0];
    const _Float16* vbuf = &lds_v[wave][cur][0];

    // ---- S(16x32) = Q(16x64) * K^T ----
    v8f c0 = {}, c1 = {};
    #pragma unroll
    for (int f = 0; f < 2; ++f) {
      v16h kb0, kb1;
      const _Float16* p0 = kbuf + (size_t)ln * DH_ + f * 32 + hlf * 16;
      const _Float16* p1 = kbuf + (size_t)(16 + ln) * DH_ + f * 32 + hlf * 16;
      #pragma unroll
      for (int e = 0; e < 16; ++e) { kb0[e] = p0[e]; kb1[e] = p1[e]; }
      c0 = wmma_f16(qa[f], kb0, c0);
      c1 = wmma_f16(qa[f], kb1, c1);
    }

    const float mk0 = mb[s0 + ln];
    const float mk1 = mb[s0 + 16 + ln];
    float sc0[8], sc1[8], mx[8];
    #pragma unroll
    for (int v = 0; v < 8; ++v) {
      sc0[v] = c0[v] + mk0;
      sc1[v] = c1[v] + mk1;
      mx[v]  = fmaxf(sc0[v], sc1[v]);
    }
    // row-wise reductions across the 16 lanes of each half via ds_swizzle
    #pragma unroll
    for (int v = 0; v < 8; ++v) {
      mx[v] = fmaxf(mx[v], swz_xor<1>(mx[v]));
      mx[v] = fmaxf(mx[v], swz_xor<2>(mx[v]));
      mx[v] = fmaxf(mx[v], swz_xor<4>(mx[v]));
      mx[v] = fmaxf(mx[v], swz_xor<8>(mx[v]));
    }
    float alpha[8], rs[8];
    #pragma unroll
    for (int v = 0; v < 8; ++v) {
      const float mn = fmaxf(m[v], mx[v]);
      alpha[v] = __expf(m[v] - mn);
      m[v] = mn;
      sc0[v] = __expf(sc0[v] - mn);
      sc1[v] = __expf(sc1[v] - mn);
      rs[v]  = sc0[v] + sc1[v];
    }
    #pragma unroll
    for (int v = 0; v < 8; ++v) {
      rs[v] += swz_xor<1>(rs[v]);
      rs[v] += swz_xor<2>(rs[v]);
      rs[v] += swz_xor<4>(rs[v]);
      rs[v] += swz_xor<8>(rs[v]);
      l[v] = l[v] * alpha[v] + rs[v];
    }
    #pragma unroll
    for (int t = 0; t < 4; ++t)
      #pragma unroll
      for (int v = 0; v < 8; ++v) accO[t][v] *= alpha[v];

    // ---- stage P (C-layout) to LDS, reload as A-fragment ----
    #pragma unroll
    for (int v = 0; v < 8; ++v) {
      const int r = v + 8 * hlf;
      lds_p[wave][r * 32 + ln]      = (_Float16)sc0[v];
      lds_p[wave][r * 32 + 16 + ln] = (_Float16)sc1[v];
    }
    asm volatile("s_wait_dscnt 0" ::: "memory");  // same-wave LDS RAW
    v16h pa;
    {
      const _Float16* pp = &lds_p[wave][ln * 32 + hlf * 16];
      #pragma unroll
      for (int j = 0; j < 8; ++j) { pa[j] = pp[j]; pa[8 + j] = pp[8 + j]; }
    }

    // ---- O(16x64) += P(16x32) * V(32x64) ----
    #pragma unroll
    for (int t = 0; t < 4; ++t) {
      v16h vb;
      const _Float16* pv = vbuf + (size_t)(t * 16 + ln) * 32 + hlf * 16;
      #pragma unroll
      for (int e = 0; e < 16; ++e) vb[e] = pv[e];
      accO[t] = wmma_f16(pa, vb, accO[t]);
    }
  }

  // ---- combine the 4 waves' (m, l, O) partials ----
  if (ln == 0) {
    #pragma unroll
    for (int v = 0; v < 8; ++v) {
      lds_m[wave][hlf * 8 + v] = m[v];
      lds_l[wave][hlf * 8 + v] = l[v];
    }
  }
  #pragma unroll
  for (int t = 0; t < 4; ++t)
    #pragma unroll
    for (int v = 0; v < 8; ++v)
      lds_o[wave][v + 8 * hlf][t * 16 + ln] = accO[t][v];
  __syncthreads();

  for (int e = threadIdx.x; e < 16 * DH_; e += 128) {
    const int row = e >> 6;
    const int d   = e & 63;
    float mf = -1e30f;
    #pragma unroll
    for (int w = 0; w < 4; ++w) mf = fmaxf(mf, lds_m[w][row]);
    float lf = 0.f, val = 0.f;
    #pragma unroll
    for (int w = 0; w < 4; ++w) {
      const float sc = __expf(lds_m[w][row] - mf);
      lf  += sc * lds_l[w][row];
      val += sc * lds_o[w][row][d];
    }
    ctx[((size_t)b * T_ + q0 + row) * HID_ + (size_t)h * DH_ + d] =
        (_Float16)(val / lf);
  }
}

// ---------------------------------------------------------------------------
extern "C" void kernel_launch(void* const* d_in, const int* in_sizes, int n_in,
                              void* d_out, int out_size, void* d_ws, size_t ws_size,
                              hipStream_t stream) {
  (void)in_sizes; (void)n_in; (void)out_size; (void)ws_size;

  const float* Xq   = (const float*)d_in[0];
  const float* Xk   = (const float*)d_in[1];
  const float* Xv   = (const float*)d_in[2];
  const float* mask = (const float*)d_in[3];
  const float* Wq   = (const float*)d_in[4];
  const float* bq   = (const float*)d_in[5];
  const float* Wk   = (const float*)d_in[6];
  const float* bk   = (const float*)d_in[7];
  const float* Wv   = (const float*)d_in[8];
  const float* bv   = (const float*)d_in[9];
  const float* Wo   = (const float*)d_in[10];
  const float* bo   = (const float*)d_in[11];

  char* ws = (char*)d_ws;
  const size_t XH = (size_t)B_ * T_ * HID_ * sizeof(_Float16);  // 8 MB
  const size_t WH = (size_t)HID_ * HID_ * sizeof(_Float16);     // 512 KB
  _Float16* xqh  = (_Float16*)(ws);
  _Float16* xkh  = (_Float16*)(ws + 1 * XH);
  _Float16* xvh  = (_Float16*)(ws + 2 * XH);
  _Float16* qhp  = (_Float16*)(ws + 3 * XH);
  _Float16* khp  = (_Float16*)(ws + 4 * XH);
  _Float16* vtp  = (_Float16*)(ws + 5 * XH);
  _Float16* ctxh = (_Float16*)(ws + 6 * XH);
  _Float16* wqh  = (_Float16*)(ws + 7 * XH);
  _Float16* wkh  = (_Float16*)(ws + 7 * XH + 1 * WH);
  _Float16* wvh  = (_Float16*)(ws + 7 * XH + 2 * WH);
  _Float16* woh  = (_Float16*)(ws + 7 * XH + 3 * WH);  // total 58 MB

  const int NX = B_ * T_ * HID_;     // 4,194,304
  const int NW = HID_ * HID_;        // 262,144
  cvt_f16_kernel<<<NX / 4 / 256, 256, 0, stream>>>(Xq, xqh, NX);
  cvt_f16_kernel<<<NX / 4 / 256, 256, 0, stream>>>(Xk, xkh, NX);
  cvt_f16_kernel<<<NX / 4 / 256, 256, 0, stream>>>(Xv, xvh, NX);
  cvt_f16_kernel<<<NW / 4 / 256, 256, 0, stream>>>(Wq, wqh, NW);
  cvt_f16_kernel<<<NW / 4 / 256, 256, 0, stream>>>(Wk, wkh, NW);
  cvt_f16_kernel<<<NW / 4 / 256, 256, 0, stream>>>(Wv, wvh, NW);
  cvt_f16_kernel<<<NW / 4 / 256, 256, 0, stream>>>(Wo, woh, NW);

  const dim3 gg(B_ * T_ / 64, HID_ / 64), gb(128);
  gemm512_kernel<1><<<gg, gb, 0, stream>>>(xqh, wqh, bq, nullptr, qhp);
  gemm512_kernel<1><<<gg, gb, 0, stream>>>(xkh, wkh, bk, nullptr, khp);
  gemm512_kernel<2><<<gg, gb, 0, stream>>>(xvh, wvh, bv, nullptr, vtp);

  flash_attn_kernel<<<dim3(T_ / 16, B_ * H_), 128, 0, stream>>>(qhp, khp, vtp, mask, ctxh);

  gemm512_kernel<0><<<gg, gb, 0, stream>>>(ctxh, woh, bo, (float*)d_out, nullptr);
}
